// LSTM_Actor_50474455662896
// MI455X (gfx1250) — matrix operand order repaired and checked by reading, not compiled, not run on hardware
//
#include <hip/hip_runtime.h>
#include <hip/hip_bf16.h>

// ---------------------------------------------------------------------------
// LSTM actor head for MI455X (gfx1250).
// Fused recurrent step (one kernel / timestep):
//   z = x_t@W_x + h@W_h (bf16 WMMA, f32 accum) -> gates -> c,h update.
// B (weight) fragments are staged per-block in double-buffered LDS (the four
// mt-waves of a block consume identical B fragments -> 4x L2 traffic cut),
// with the K-loop split into branch-free, fully unrolled x- and h-phases so
// buffer parity and all ds/global offsets are compile-time immediates.
// Gate biases are pre-folded into the WMMA accumulator init (per-lane splat).
// h is double-buffered in packed WMMA A-fragment layout across steps.
// ---------------------------------------------------------------------------

typedef __attribute__((ext_vector_type(16))) __bf16 v16bf;
typedef __attribute__((ext_vector_type(8)))  float  v8f;

#define Bsz   64
#define Tsz   256
#define Ssz   512
#define Hsz   1024
#define G4H   4096   // 4*H
#define HD    512    // H/2
#define Asz   64
#define NOUT  64
#define KT_X  16     // k-tiles from x (512/32)
#define KT_H  32     // k-tiles from h (1024/32)

// ---------------- fragment index helpers (CDNA5 ISA 7.12.2) ----------------
// A-matrix (16x32 bf16): k_local = (v/4)*16 + (lane/16)*8 + (v%4)*2 + e
__device__ __forceinline__ size_t a_frag_pos(int row, int k, int KT) {
  int mt  = row >> 4, rl = row & 15;
  int kt  = k >> 5,   a  = k & 31;
  int e   = a & 1;
  int vm  = (a >> 1) & 3;
  int grp = (a >> 3) & 1;
  int vh  = (a >> 4) & 1;
  int v   = vm + (vh << 2);
  int lane = rl + (grp << 4);
  return ((size_t)(mt * KT + kt) * 32 + lane) * 16 + (v * 2 + e);
}

// B-matrix (32x16 bf16): k_local = (v/4)*16 + (lane/16)*8 + (v%4)*2 + e (same as A); N=lane%16.
__device__ __forceinline__ size_t b_frag_pos(int k, int n, int NT) {
  int lane = (n & 15) | (((k >> 3) & 1) << 4);
  return ((size_t)((k >> 5) * NT + (n >> 4)) * 32 + lane) * 16 + ((k & 7) | (((k >> 4) & 1) << 3));
}

__device__ __forceinline__ float sigf(float x) {
  return 1.0f / (1.0f + __expf(-x));
}

// 4 gate-WMMAs from one A fragment and 4 LDS-resident B fragments.
__device__ __forceinline__ void mac4(v16bf a, const v16bf* rd,
                                     v8f& accI, v8f& accF, v8f& accG, v8f& accO) {
  v16bf bi = rd[0 * 32];
  v16bf bf = rd[1 * 32];
  v16bf bg = rd[2 * 32];
  v16bf bo = rd[3 * 32];
  accI = __builtin_amdgcn_wmma_f32_16x16x32_bf16(false, a, false, bi, (short)0, accI, false, false);
  accF = __builtin_amdgcn_wmma_f32_16x16x32_bf16(false, a, false, bf, (short)0, accF, false, false);
  accG = __builtin_amdgcn_wmma_f32_16x16x32_bf16(false, a, false, bg, (short)0, accG, false, false);
  accO = __builtin_amdgcn_wmma_f32_16x16x32_bf16(false, a, false, bo, (short)0, accO, false, false);
}

// ---------------- prep kernels ----------------
__global__ void pack_b_kernel(const float* __restrict__ W, __bf16* __restrict__ dst,
                              int K, int N) {
  int idx = blockIdx.x * blockDim.x + threadIdx.x;
  if (idx >= K * N) return;
  int k = idx / N, n = idx - k * N;
  dst[b_frag_pos(k, n, N >> 4)] = (__bf16)W[idx];
}

// x:[B,T,S] -> per-timestep A-frag slabs (each slab: 64 x 512 A matrix).
__global__ void pack_x_kernel(const float* __restrict__ x, __bf16* __restrict__ dst) {
  int idx = blockIdx.x * blockDim.x + threadIdx.x;
  if (idx >= Bsz * Tsz * Ssz) return;
  int b = idx / (Tsz * Ssz);
  int t = (idx / Ssz) % Tsz;
  int s = idx & (Ssz - 1);
  // slab stride: 4 mtiles * 16 ktiles * 32 lanes * 16 elems = 32768 bf16
  dst[(size_t)t * 32768 + a_frag_pos(b, s, Ssz / 32)] = (__bf16)x[idx];
}

__global__ void init_state_kernel(float* __restrict__ c, __bf16* __restrict__ hp0) {
  int idx = blockIdx.x * blockDim.x + threadIdx.x;
  if (idx >= Bsz * Hsz) return;
  c[idx]   = 0.0f;
  hp0[idx] = (__bf16)0.0f;
}

// ---------------- fused recurrent step ----------------
// Grid: 32 blocks x 256 threads (8 waves). Wave wl: mt = wl&3, ntcL = wl>>2.
// Block covers hidden-column tiles ntc = blockIdx*2 + {0,1}; each wave computes
// the 4 gate-aligned tiles (i,f,g,o) for (mt, ntc) -> in-register gate epilogue.
// Thread (wl,lane) stages B fragment f = wl, lane `lane` of each kt-slice.
__global__ __launch_bounds__(256)
void lstm_step_fused_kernel(const __bf16* __restrict__ xp,
                            const __bf16* __restrict__ hin,
                            const __bf16* __restrict__ wxp,
                            const __bf16* __restrict__ whp,
                            const float*  __restrict__ bias,
                            float*        __restrict__ c,
                            __bf16*       __restrict__ hout,
                            int t) {
  __shared__ __bf16 sB[2][8][32][16];   // [buf][frag][lane][elem] = 16 KB

  const int tid  = threadIdx.x;
  const int lane = tid & 31;
  const int wl   = tid >> 5;                 // 0..7
  const int mt   = wl & 3;
  const int ntcL = wl >> 2;                  // 0..1
  const int ntc  = blockIdx.x * 2 + ntcL;    // 0..63 hidden-column tile

  const v16bf* xA  = (const v16bf*)xp + (size_t)t * 2048; // 2048 v16bf per slab
  const v16bf* hA  = (const v16bf*)hin;
  const v16bf* wxB = (const v16bf*)wxp;
  const v16bf* whB = (const v16bf*)whp;

  // staging role: this thread stages frag f = wl (= ntcS*4 + gate), lane `lane`
  const int ntS = (blockIdx.x * 2 + (wl >> 2)) + 64 * (wl & 3); // n-tile of frag wl

  // per-thread LDS write/read pointers (frag stride = 32 v16bf)
  v16bf*       wr0 = (v16bf*)&sB[0][wl][lane][0];
  v16bf*       wr1 = (v16bf*)&sB[1][wl][lane][0];
  const v16bf* rd0 = (const v16bf*)&sB[0][ntcL * 4][lane][0];
  const v16bf* rd1 = (const v16bf*)&sB[1][ntcL * 4][lane][0];

  // fold gate biases into accumulator init: D-frag column is per-lane constant
  const int n  = ntc * 16 + (lane & 15);
  const float bI = bias[n];
  const float bF = bias[n + Hsz];
  const float bG = bias[n + 2 * Hsz];
  const float bO = bias[n + 3 * Hsz];
  v8f accI = {bI, bI, bI, bI, bI, bI, bI, bI};
  v8f accF = {bF, bF, bF, bF, bF, bF, bF, bF};
  v8f accG = {bG, bG, bG, bG, bG, bG, bG, bG};
  v8f accO = {bO, bO, bO, bO, bO, bO, bO, bO};

  // prologue: stage slice 0 (from W_x) into buf 0
  *wr0 = wxB[((size_t)0 * 256 + ntS) * 32 + lane];

  // ---- x phase: global slice kt staged in buf (kt&1); stage kt+1 ahead ----
#pragma unroll
  for (int kt = 0; kt < KT_X; ++kt) {
    __syncthreads();
    // next slice source: wx[kt+1], or wh[0] at the phase crossover (kt==15)
    const v16bf* src = (kt + 1 < KT_X)
        ? (wxB + ((size_t)(kt + 1) * 256 + ntS) * 32 + lane)
        : (whB + ((size_t)ntS) * 32 + lane);
    __builtin_prefetch(src + (size_t)256 * 32, 0, 1);
    v16bf s = *src;
    *((kt & 1) ? wr0 : wr1) = s;                       // into buf (kt+1)&1
    v16bf a = xA[((size_t)(mt * KT_X + kt)) * 32 + lane];
    mac4(a, (kt & 1) ? rd1 : rd0, accI, accF, accG, accO);
  }

  // ---- h phase: global index 16+kt -> buf parity is again (kt&1) ----
#pragma unroll
  for (int kt = 0; kt < KT_H; ++kt) {
    __syncthreads();
    if (kt + 1 < KT_H) {
      const v16bf* src = whB + ((size_t)(kt + 1) * 256 + ntS) * 32 + lane;
      __builtin_prefetch(src + (size_t)256 * 32, 0, 1);
      v16bf s = *src;
      *((kt & 1) ? wr0 : wr1) = s;
    }
    v16bf a = hA[((size_t)(mt * KT_H + kt)) * 32 + lane];
    mac4(a, (kt & 1) ? rd1 : rd0, accI, accF, accG, accO);
  }

  // ---- in-register LSTM gate epilogue ----
  // D layout: lanes 0-15 -> rows mt*16 + r, lanes 16-31 -> rows mt*16+8+r.
  const int mbase = mt * 16 + ((lane >> 4) << 3);
#pragma unroll
  for (int r = 0; r < 8; ++r) {
    const int row = mbase + r;
    const size_t ci = (size_t)row * Hsz + n;
    float gi = sigf(accI[r]);
    float gf = sigf(accF[r]);
    float gg = tanhf(accG[r]);
    float go = sigf(accO[r]);
    float cn = gf * c[ci] + gi * gg;
    c[ci] = cn;
    float h = go * tanhf(cn);
    hout[a_frag_pos(row, n, Hsz / 32)] = (__bf16)h;  // packed A-frag for next step
  }
}

// ---------------- head: hid = relu(h @ Wd1 + bd1) ----------------
// M=64 (4 mt), N=512 (32 nt), K=1024 (32 kt). 32 waves, 4 tiles each.
__global__ void hid_gemm_kernel(const __bf16* __restrict__ hp,
                                const __bf16* __restrict__ wdp,
                                const float* __restrict__ bd1,
                                float* __restrict__ hid) {
  const int lane = threadIdx.x & 31;
  const int w    = blockIdx.x * (blockDim.x >> 5) + (threadIdx.x >> 5); // 0..31
  const int mt   = w & 3;
  const int ntb  = (w >> 2) << 2;

  const v16bf* hA = (const v16bf*)hp;
  const v16bf* wB = (const v16bf*)wdp;

  v8f acc0 = {}, acc1 = {}, acc2 = {}, acc3 = {};
  for (int kt = 0; kt < 32; ++kt) {
    v16bf a = hA[((size_t)(mt * 32 + kt)) * 32 + lane];
    const v16bf* bb = wB + ((size_t)kt * 32) * 32 + lane;
    v16bf b0 = bb[(size_t)(ntb + 0) * 32];
    v16bf b1 = bb[(size_t)(ntb + 1) * 32];
    v16bf b2 = bb[(size_t)(ntb + 2) * 32];
    v16bf b3 = bb[(size_t)(ntb + 3) * 32];
    acc0 = __builtin_amdgcn_wmma_f32_16x16x32_bf16(false, a, false, b0, (short)0, acc0, false, false);
    acc1 = __builtin_amdgcn_wmma_f32_16x16x32_bf16(false, a, false, b1, (short)0, acc1, false, false);
    acc2 = __builtin_amdgcn_wmma_f32_16x16x32_bf16(false, a, false, b2, (short)0, acc2, false, false);
    acc3 = __builtin_amdgcn_wmma_f32_16x16x32_bf16(false, a, false, b3, (short)0, acc3, false, false);
  }

  const int mbase = mt * 16 + ((lane >> 4) << 3);
  const int ncol  = lane & 15;
#pragma unroll
  for (int r = 0; r < 8; ++r) {
    int n0 = (ntb + 0) * 16 + ncol;
    int n1 = (ntb + 1) * 16 + ncol;
    int n2 = (ntb + 2) * 16 + ncol;
    int n3 = (ntb + 3) * 16 + ncol;
    hid[(size_t)(mbase + r) * HD + n0] = fmaxf(acc0[r] + bd1[n0], 0.0f);
    hid[(size_t)(mbase + r) * HD + n1] = fmaxf(acc1[r] + bd1[n1], 0.0f);
    hid[(size_t)(mbase + r) * HD + n2] = fmaxf(acc2[r] + bd1[n2], 0.0f);
    hid[(size_t)(mbase + r) * HD + n3] = fmaxf(acc3[r] + bd1[n3], 0.0f);
  }
}

// ---------------- output: feat=[hid|actions|horizon] @ Wd2 + bd2 ----------------
__global__ void out_kernel(const float* __restrict__ hid,
                           const float* __restrict__ actions,
                           const float* __restrict__ horizon,
                           const float* __restrict__ Wd2,
                           const float* __restrict__ bd2,
                           float* __restrict__ out) {
  int row = blockIdx.x;     // 0..63
  int j   = threadIdx.x;    // 0..63
  float s = bd2[j];
  const float* hr = hid + (size_t)row * HD;
  for (int k = 0; k < HD; ++k)  s += hr[k] * Wd2[(size_t)k * NOUT + j];
  const float* ar = actions + (size_t)row * Asz;
  for (int k = 0; k < Asz; ++k) s += ar[k] * Wd2[(size_t)(HD + k) * NOUT + j];
  s += horizon[row] * Wd2[(size_t)(HD + Asz) * NOUT + j];
  out[(size_t)row * NOUT + j] = s;
}

// ---------------------------------------------------------------------------
extern "C" void kernel_launch(void* const* d_in, const int* in_sizes, int n_in,
                              void* d_out, int out_size, void* d_ws, size_t ws_size,
                              hipStream_t stream) {
  const float* x       = (const float*)d_in[0];
  const float* actions = (const float*)d_in[1];
  const float* horizon = (const float*)d_in[2];
  const float* W_x     = (const float*)d_in[3];
  const float* W_h     = (const float*)d_in[4];
  const float* bias    = (const float*)d_in[5];
  const float* Wd1     = (const float*)d_in[6];
  const float* bd1     = (const float*)d_in[7];
  const float* Wd2     = (const float*)d_in[8];
  const float* bd2     = (const float*)d_in[9];
  float* out = (float*)d_out;

  char* ws = (char*)d_ws;
  // workspace layout (all 32B aligned)
  float*  c    = (float*)(ws + 0x000000);    // 64*1024*4   = 256 KB
  __bf16* hp0  = (__bf16*)(ws + 0x040000);   // 64*1024*2   = 128 KB
  __bf16* hp1  = (__bf16*)(ws + 0x060000);   // 64*1024*2   = 128 KB
  float*  hid  = (float*)(ws + 0x080000);    // 64*512*4    = 128 KB
  __bf16* wxp  = (__bf16*)(ws + 0x0A0000);   // 512*4096*2  = 4 MB
  __bf16* whp  = (__bf16*)(ws + 0x4A0000);   // 1024*4096*2 = 8 MB
  __bf16* wd1p = (__bf16*)(ws + 0xCA0000);   // 1024*512*2  = 1 MB
  __bf16* xp   = (__bf16*)(ws + 0xDA0000);   // 16384*512*2 = 16 MB

  // ---- prep (deterministic, every call) ----
  pack_b_kernel<<<(Ssz * G4H + 255) / 256, 256, 0, stream>>>(W_x, wxp, Ssz, G4H);
  pack_b_kernel<<<(Hsz * G4H + 255) / 256, 256, 0, stream>>>(W_h, whp, Hsz, G4H);
  pack_b_kernel<<<(Hsz * HD  + 255) / 256, 256, 0, stream>>>(Wd1, wd1p, Hsz, HD);
  pack_x_kernel<<<(Bsz * Tsz * Ssz + 255) / 256, 256, 0, stream>>>(x, xp);
  init_state_kernel<<<(Bsz * Hsz + 255) / 256, 256, 0, stream>>>(c, hp0);

  // ---- 256 fused recurrent steps (h ping-pongs hp0 <-> hp1) ----
  for (int t = 0; t < Tsz; ++t) {
    const __bf16* hin  = (t & 1) ? hp1 : hp0;
    __bf16*       hout = (t & 1) ? hp0 : hp1;
    lstm_step_fused_kernel<<<32, 256, 0, stream>>>(xp, hin, wxp, whp, bias, c, hout, t);
  }
  // t=255 (odd) wrote hp0 -> final h lives in hp0.

  // ---- head ----
  hid_gemm_kernel<<<4, 256, 0, stream>>>(hp0, wd1p, bd1, hid);
  out_kernel<<<Bsz, NOUT, 0, stream>>>(hid, actions, horizon, Wd2, bd2, out);
}
